// MultiHeadSelfAttention_6124623364360
// MI455X (gfx1250) — matrix operand order repaired and checked by reading, not compiled, not run on hardware
//
#include <hip/hip_runtime.h>
#include <hip/hip_bf16.h>
#include <stdint.h>

// MI455X / gfx1250, wave32. WMMA f16 GEMMs + TDM (tensor_load_to_lds) bias prefetch.
// B=4, S=1024, D=1024, H=16, HD=64

typedef __attribute__((ext_vector_type(16))) _Float16 v16h;
typedef __attribute__((ext_vector_type(8)))  float    v8f;
typedef __attribute__((ext_vector_type(4)))  unsigned int v4u;
typedef __attribute__((ext_vector_type(8)))  int      v8i;
typedef __attribute__((ext_vector_type(4)))  int      v4i;

#define WMMA_F16(a, b, c) \
  __builtin_amdgcn_wmma_f32_16x16x32_f16(false, (a), false, (b), (short)0, (c), false, false)

static __device__ __forceinline__ v8f zero8() {
  v8f z;
#pragma unroll
  for (int i = 0; i < 8; ++i) z[i] = 0.0f;
  return z;
}

// A fragment: 16x32 f16 tile at (row0, k0) from row-major [*, ld].
static __device__ __forceinline__ v16h load_a16(const _Float16* base, int ld, int row0, int k0, int lane) {
  const int M = lane & 15, hi = lane >> 4;
  const _Float16* p = base + (size_t)(row0 + M) * ld + k0 + hi * 8;
  v16h a;
  *reinterpret_cast<uint4*>(&a)       = *reinterpret_cast<const uint4*>(p);
  *(reinterpret_cast<uint4*>(&a) + 1) = *reinterpret_cast<const uint4*>(p + 16);
  return a;
}

// B fragment: 32x16 f16 tile; element (k, n) read from base[(n0+n)*ld + k0 + k]
static __device__ __forceinline__ v16h load_b16(const _Float16* base, int ld, int n0, int k0, int lane) {
  const int N = lane & 15, hi = lane >> 4;
  const _Float16* p = base + (size_t)(n0 + N) * ld + k0 + hi * 8;
  v16h b;
  *reinterpret_cast<uint4*>(&b)       = *reinterpret_cast<const uint4*>(p);
  *(reinterpret_cast<uint4*>(&b) + 1) = *reinterpret_cast<const uint4*>(p + 16);
  return b;
}

// TDM: DMA a 16x32 fp32 tile (row stride 1024 floats) from global to LDS.
// D# built per CDNA5 ISA.
static __device__ __forceinline__ void tdm_load_bias_tile(const float* gptr, unsigned int lds_byte_off) {
  const uint64_t ga = (uint64_t)(uintptr_t)gptr;
  v4u g0 = {0u, 0u, 0u, 0u};
  g0[0] = 1u;                                            // count=1, load, user mode
  g0[1] = lds_byte_off;                                  // lds_addr (bytes)
  g0[2] = (unsigned int)(ga & 0xFFFFFFFFu);              // global_addr[31:0]
  g0[3] = (unsigned int)((ga >> 32) & 0x1FFFFFFu) | (2u << 30);  // [56:32] + type=2
  const unsigned int td0 = 1u << 30;                     // tensor_dim0 (huge, no OOB)
  const unsigned int td1 = 1u << 20;                     // tensor_dim1 (huge)
  v8i g1 = {0, 0, 0, 0, 0, 0, 0, 0};
  g1[0] = (int)(2u << 16);                               // mask=0, data_size=2 (4B)
  g1[1] = (int)((td0 & 0xFFFFu) << 16);                  // [63:48] = td0 lo16
  g1[2] = (int)((td0 >> 16) | ((td1 & 0xFFFFu) << 16));  // td0 hi16 | td1 lo16
  g1[3] = (int)((td1 >> 16) | (32u << 16));              // td1 hi16 | tile_dim0=32
  g1[4] = (int)16u;                                      // tile_dim1=16, tile_dim2=0
  g1[5] = (int)1024u;                                    // tensor_dim0_stride = 1024
  v4i g2 = {0, 0, 0, 0};
  v4i g3 = {0, 0, 0, 0};
#if defined(__clang_major__) && (__clang_major__ >= 23)
  v8i g4 = {0, 0, 0, 0, 0, 0, 0, 0};
  __builtin_amdgcn_tensor_load_to_lds(g0, g1, g2, g3, g4, 0);
#else
  __builtin_amdgcn_tensor_load_to_lds(g0, g1, g2, g3, 0);
#endif
}

__global__ void cvt_kernel(const float* __restrict__ src, _Float16* __restrict__ dst, int n) {
  int i = blockIdx.x * blockDim.x + threadIdx.x;
  if (i < n) dst[i] = (_Float16)src[i];
}

// Fused QKV projection + bias + RoPE. One wave = 32 rows x 64 cols (one head-slice).
// Ping-pong k-pipeline (step 64 = two 32-wide phases), no cross-buffer copies.
__global__ void __launch_bounds__(256) qkv_rope_kernel(
    const _Float16* __restrict__ xh, const _Float16* __restrict__ wqkv,
    const float* __restrict__ bq, const float* __restrict__ bk, const float* __restrict__ bv,
    const int* __restrict__ pos_ids,
    _Float16* __restrict__ qb, _Float16* __restrict__ kb, _Float16* __restrict__ vt) {
  const int lane = threadIdx.x & 31;
  const int wid  = (blockIdx.x * blockDim.x + threadIdx.x) >> 5;
  const int g    = wid % 48;
  const int mt   = wid / 48;        // 128 tiles of 32 rows
  const int row0 = mt * 32;
  const int n0   = g * 64;
  const int N = lane & 15, hi = lane >> 4;

  v8f acc[2][4];
#pragma unroll
  for (int m = 0; m < 2; ++m)
#pragma unroll
    for (int t = 0; t < 4; ++t) acc[m][t] = zero8();

  v16h aA[2], bA[4], aB[2], bB[4];
  aA[0] = load_a16(xh, 1024, row0, 0, lane);
  aA[1] = load_a16(xh, 1024, row0 + 16, 0, lane);
#pragma unroll
  for (int t = 0; t < 4; ++t) bA[t] = load_b16(wqkv, 1024, n0 + t * 16, 0, lane);

  for (int k0 = 0; k0 < 1024; k0 += 64) {
    // phase-B prefetch (k0+32 always in range)
    aB[0] = load_a16(xh, 1024, row0, k0 + 32, lane);
    aB[1] = load_a16(xh, 1024, row0 + 16, k0 + 32, lane);
#pragma unroll
    for (int t = 0; t < 4; ++t) bB[t] = load_b16(wqkv, 1024, n0 + t * 16, k0 + 32, lane);
    // compute phase A
#pragma unroll
    for (int t = 0; t < 4; ++t) {
      acc[0][t] = WMMA_F16(aA[0], bA[t], acc[0][t]);
      acc[1][t] = WMMA_F16(aA[1], bA[t], acc[1][t]);
    }
    // phase-A prefetch for next iteration
    if (k0 + 64 < 1024) {
      aA[0] = load_a16(xh, 1024, row0, k0 + 64, lane);
      aA[1] = load_a16(xh, 1024, row0 + 16, k0 + 64, lane);
#pragma unroll
      for (int t = 0; t < 4; ++t) bA[t] = load_b16(wqkv, 1024, n0 + t * 16, k0 + 64, lane);
    }
    // compute phase B
#pragma unroll
    for (int t = 0; t < 4; ++t) {
      acc[0][t] = WMMA_F16(aB[0], bB[t], acc[0][t]);
      acc[1][t] = WMMA_F16(aB[1], bB[t], acc[1][t]);
    }
  }

  const int type = g >> 4;   // 0=Q,1=K,2=V
  const int h    = g & 15;
  const float* bias = (type == 0) ? bq : (type == 1) ? bk : bv;
#pragma unroll
  for (int t = 0; t < 4; ++t) {
    float bb = bias[h * 64 + t * 16 + N];
#pragma unroll
    for (int m = 0; m < 2; ++m)
#pragma unroll
      for (int j = 0; j < 8; ++j) acc[m][t][j] += bb;
  }

  if (type == 2) {  // V -> transposed [B,H,HD,S] f16
#pragma unroll
    for (int m = 0; m < 2; ++m)
#pragma unroll
      for (int j = 0; j < 8; ++j) {
        int row = row0 + m * 16 + j + 8 * hi;
        int b = row >> 10, s = row & 1023;
#pragma unroll
        for (int t = 0; t < 4; ++t) {
          int hd = t * 16 + N;
          vt[((size_t)(b * 16 + h) * 64 + hd) * 1024 + s] = (_Float16)acc[m][t][j];
        }
      }
  } else {  // Q/K: RoPE then [B,H,S,HD] f16
    _Float16* dst = (type == 0) ? qb : kb;
    const float k0f  = -9.210340371976184f / 32.0f;  // -ln(10000)/32
    const float inv0 = __expf(k0f * (float)N);
    const float inv1 = __expf(k0f * (float)(N + 16));
#pragma unroll
    for (int m = 0; m < 2; ++m)
#pragma unroll
      for (int j = 0; j < 8; ++j) {
        int row = row0 + m * 16 + j + 8 * hi;
        int b = row >> 10, s = row & 1023;
        float pos = (float)pos_ids[b * 1024 + s];
        float th0 = pos * inv0, th1 = pos * inv1;
        float c0 = __cosf(th0), s0 = __sinf(th0);
        float c1 = __cosf(th1), s1 = __sinf(th1);
#pragma unroll
        for (int t = 0; t < 4; ++t) {
          float cs = (t & 1) ? c1 : c0;
          float sn = (t & 1) ? s1 : s0;
          float rot = (t < 2) ? -acc[m][t + 2][j] : acc[m][t - 2][j];
          float val = acc[m][t][j] * cs + rot * sn;
          int hd = t * 16 + N;
          dst[((size_t)(b * 16 + h) * 1024 + s) * 64 + hd] = (_Float16)val;
        }
      }
  }
}

// Flash attention: one wave per (b, h, 16-query tile); 32 keys per step.
// attn_bias tiles are DMA'd by the TDM into double-buffered LDS one step ahead.
__global__ void __launch_bounds__(256) attn_kernel(
    const _Float16* __restrict__ qb, const _Float16* __restrict__ kbuf,
    const _Float16* __restrict__ vt, const float* __restrict__ attn_bias,
    const unsigned char* __restrict__ pad_mask, _Float16* __restrict__ ctx) {
  __shared__ __align__(16) _Float16 plds[8 * 16 * 40];     // per-wave 16x32 P tile (stride 40)
  __shared__ __align__(16) float    biasLds[8][2][16 * 32];// per-wave double-buffered bias tile
  const int lane = threadIdx.x & 31;
  const int wv   = threadIdx.x >> 5;
  const int wid  = (blockIdx.x * blockDim.x + threadIdx.x) >> 5;
  const int qt = wid & 63;
  const int h  = (wid >> 6) & 15;
  const int b  = wid >> 10;
  const int N = lane & 15, hi = lane >> 4;
  _Float16* myp  = plds + wv * (16 * 40);
  float*    myBias = &biasLds[wv][0][0];

  const _Float16* qbh = qb   + (size_t)(b * 16 + h) * 1024 * 64;
  const _Float16* kbh = kbuf + (size_t)(b * 16 + h) * 1024 * 64;
  const _Float16* vth = vt   + (size_t)(b * 16 + h) * 64 * 1024;
  const float* biasq  = attn_bias + (size_t)(b * 16 + h) * 1024 * 1024 + (size_t)qt * 16 * 1024;
  const unsigned char* pm = pad_mask + b * 1024;

  const v16h aq0 = load_a16(qbh, 64, qt * 16, 0, lane);
  const v16h aq1 = load_a16(qbh, 64, qt * 16, 32, lane);

  v8f acc[4] = {zero8(), zero8(), zero8(), zero8()};
  float mrun[8], lrun[8];
#pragma unroll
  for (int j = 0; j < 8; ++j) { mrun[j] = -3.0e38f; lrun[j] = 0.0f; }

  // prefetch first bias tile
  tdm_load_bias_tile(biasq, (unsigned int)(uintptr_t)myBias);

  for (int kbase = 0; kbase < 1024; kbase += 32) {
    const int cur = (kbase >> 5) & 1;
    const bool has_next = (kbase + 32) < 1024;
    if (has_next)
      tdm_load_bias_tile(biasq + kbase + 32,
                         (unsigned int)(uintptr_t)(myBias + (cur ^ 1) * 512));

    // ---- scores: S[16x32] = Q[16x64] @ K_chunk^T ----
    v8f s0 = zero8(), s1 = zero8();
    {
      v16h b00 = load_b16(kbh, 64, kbase,      0,  lane);
      v16h b01 = load_b16(kbh, 64, kbase,      32, lane);
      v16h b10 = load_b16(kbh, 64, kbase + 16, 0,  lane);
      v16h b11 = load_b16(kbh, 64, kbase + 16, 32, lane);
      s0 = WMMA_F16(aq0, b00, s0);
      s0 = WMMA_F16(aq1, b01, s0);
      s1 = WMMA_F16(aq0, b10, s1);
      s1 = WMMA_F16(aq1, b11, s1);
    }
    // V fragments in flight while softmax math runs
    v16h bv[4];
#pragma unroll
    for (int t = 0; t < 4; ++t) bv[t] = load_b16(vth, 1024, t * 16, kbase, lane);

    // wait for current bias tile DMA (in-order TENSORcnt)
    if (has_next) __builtin_amdgcn_s_wait_tensorcnt(1);
    else          __builtin_amdgcn_s_wait_tensorcnt(0);
    asm volatile("" ::: "memory");
    const float* bt = myBias + cur * 512;

    const bool pad0 = pm[kbase + N] != 0;
    const bool pad1 = pm[kbase + 16 + N] != 0;
#pragma unroll
    for (int j = 0; j < 8; ++j) {
      int r = j + 8 * hi;
      float x0 = s0[j] * 0.125f + bt[r * 32 + N];
      float x1 = s1[j] * 0.125f + bt[r * 32 + 16 + N];
      x0 = pad0 ? -3.0e38f : x0;
      x1 = pad1 ? -3.0e38f : x1;
      float mx = fmaxf(x0, x1);
#pragma unroll
      for (int d = 8; d >= 1; d >>= 1) mx = fmaxf(mx, __shfl_xor(mx, d, 16));
      float mnew  = fmaxf(mrun[j], mx);
      float alpha = __expf(mrun[j] - mnew);
      float p0 = __expf(x0 - mnew);
      float p1 = __expf(x1 - mnew);
      float rs = p0 + p1;
#pragma unroll
      for (int d = 8; d >= 1; d >>= 1) rs += __shfl_xor(rs, d, 16);
      lrun[j] = lrun[j] * alpha + rs;
      mrun[j] = mnew;
#pragma unroll
      for (int t = 0; t < 4; ++t) acc[t][j] *= alpha;
      // C-layout -> LDS row-major P tile (A-layout reload below)
      myp[r * 40 + N]      = (_Float16)p0;
      myp[r * 40 + 16 + N] = (_Float16)p1;
    }
    asm volatile("s_wait_dscnt 0" ::: "memory");  // cross-lane LDS RAW inside the wave
    v16h pA = load_a16(myp, 40, 0, 0, lane);
#pragma unroll
    for (int t = 0; t < 4; ++t) acc[t] = WMMA_F16(pA, bv[t], acc[t]);
  }

#pragma unroll
  for (int j = 0; j < 8; ++j) {
    float inv = 1.0f / lrun[j];
    int qrow = qt * 16 + j + 8 * hi;
    size_t orow = ((size_t)b * 1024 + qrow) * 1024 + h * 64;
#pragma unroll
    for (int t = 0; t < 4; ++t)
      ctx[orow + t * 16 + N] = (_Float16)(acc[t][j] * inv);
  }
}

// metric[b,s,hd] = mean over heads of rope(k)
__global__ void metric_kernel(const _Float16* __restrict__ kbuf, float* __restrict__ out) {
  int i = blockIdx.x * blockDim.x + threadIdx.x;  // 4*1024*64
  int b = i >> 16, s = (i >> 6) & 1023, hd = i & 63;
  float sum = 0.0f;
#pragma unroll
  for (int h = 0; h < 16; ++h)
    sum += (float)kbuf[((size_t)(b * 16 + h) * 1024 + s) * 64 + hd];
  out[i] = sum * 0.0625f;
}

// Output projection: out = ctx @ Wo^T + bo (fp32 out), 32x64 per wave, ping-pong pipeline.
__global__ void __launch_bounds__(256) oproj_kernel(
    const _Float16* __restrict__ ctx, const _Float16* __restrict__ woh,
    const float* __restrict__ bo, float* __restrict__ out) {
  const int lane = threadIdx.x & 31;
  const int wid  = (blockIdx.x * blockDim.x + threadIdx.x) >> 5;
  const int ng   = wid & 15;
  const int mt   = wid >> 4;        // 128 tiles of 32 rows
  const int row0 = mt * 32;
  const int n0   = ng * 64;
  const int N = lane & 15, hi = lane >> 4;

  v8f acc[2][4];
#pragma unroll
  for (int m = 0; m < 2; ++m)
#pragma unroll
    for (int t = 0; t < 4; ++t) acc[m][t] = zero8();

  v16h aA[2], bA[4], aB[2], bB[4];
  aA[0] = load_a16(ctx, 1024, row0, 0, lane);
  aA[1] = load_a16(ctx, 1024, row0 + 16, 0, lane);
#pragma unroll
  for (int t = 0; t < 4; ++t) bA[t] = load_b16(woh, 1024, n0 + t * 16, 0, lane);

  for (int k0 = 0; k0 < 1024; k0 += 64) {
    aB[0] = load_a16(ctx, 1024, row0, k0 + 32, lane);
    aB[1] = load_a16(ctx, 1024, row0 + 16, k0 + 32, lane);
#pragma unroll
    for (int t = 0; t < 4; ++t) bB[t] = load_b16(woh, 1024, n0 + t * 16, k0 + 32, lane);
#pragma unroll
    for (int t = 0; t < 4; ++t) {
      acc[0][t] = WMMA_F16(aA[0], bA[t], acc[0][t]);
      acc[1][t] = WMMA_F16(aA[1], bA[t], acc[1][t]);
    }
    if (k0 + 64 < 1024) {
      aA[0] = load_a16(ctx, 1024, row0, k0 + 64, lane);
      aA[1] = load_a16(ctx, 1024, row0 + 16, k0 + 64, lane);
#pragma unroll
      for (int t = 0; t < 4; ++t) bA[t] = load_b16(woh, 1024, n0 + t * 16, k0 + 64, lane);
    }
#pragma unroll
    for (int t = 0; t < 4; ++t) {
      acc[0][t] = WMMA_F16(aB[0], bB[t], acc[0][t]);
      acc[1][t] = WMMA_F16(aB[1], bB[t], acc[1][t]);
    }
  }

#pragma unroll
  for (int t = 0; t < 4; ++t) {
    int col = n0 + t * 16 + N;
    float bb = bo[col];
#pragma unroll
    for (int m = 0; m < 2; ++m)
#pragma unroll
      for (int j = 0; j < 8; ++j)
        out[(size_t)(row0 + m * 16 + j + 8 * hi) * 1024 + col] = acc[m][t][j] + bb;
  }
}

extern "C" void kernel_launch(void* const* d_in, const int* in_sizes, int n_in,
                              void* d_out, int out_size, void* d_ws, size_t ws_size,
                              hipStream_t stream) {
  (void)in_sizes; (void)n_in; (void)out_size; (void)ws_size;
  const float* x          = (const float*)d_in[0];
  const float* attn_bias  = (const float*)d_in[1];
  const int* pos_ids      = (const int*)d_in[2];
  const unsigned char* pm = (const unsigned char*)d_in[3];
  const float* Wq = (const float*)d_in[4];  const float* bq = (const float*)d_in[5];
  const float* Wk = (const float*)d_in[6];  const float* bk = (const float*)d_in[7];
  const float* Wv = (const float*)d_in[8];  const float* bv = (const float*)d_in[9];
  const float* Wo = (const float*)d_in[10]; const float* bo = (const float*)d_in[11];

  char* ws = (char*)d_ws;
  _Float16* xh   = (_Float16*)(ws + 0);          // 8 MB  x in f16
  _Float16* wqkv = (_Float16*)(ws + 8388608);    // 6 MB  [Wq;Wk;Wv] f16
  _Float16* woh  = (_Float16*)(ws + 14680064);   // 2 MB  Wo f16
  _Float16* qbuf = (_Float16*)(ws + 16777216);   // 8 MB  q rope [B,H,S,HD]
  _Float16* kbuf = (_Float16*)(ws + 25165824);   // 8 MB  k rope [B,H,S,HD]
  _Float16* vtb  = (_Float16*)(ws + 33554432);   // 8 MB  v transposed [B,H,HD,S]
  _Float16* ctx  = (_Float16*)(ws + 41943040);   // 8 MB  attention context [B*S, D]

  float* out    = (float*)d_out;
  float* metric = out + 4194304;  // 4*1024*1024 then 4*1024*64

  cvt_kernel<<<16384, 256, 0, stream>>>(x,  xh,             4194304);
  cvt_kernel<<<4096,  256, 0, stream>>>(Wq, wqkv,           1048576);
  cvt_kernel<<<4096,  256, 0, stream>>>(Wk, wqkv + 1048576, 1048576);
  cvt_kernel<<<4096,  256, 0, stream>>>(Wv, wqkv + 2097152, 1048576);
  cvt_kernel<<<4096,  256, 0, stream>>>(Wo, woh,            1048576);

  qkv_rope_kernel<<<768, 256, 0, stream>>>(xh, wqkv, bq, bk, bv, pos_ids, qbuf, kbuf, vtb);
  attn_kernel<<<512, 256, 0, stream>>>(qbuf, kbuf, vtb, attn_bias, pm, ctx);
  metric_kernel<<<1024, 256, 0, stream>>>(kbuf, metric);
  oproj_kernel<<<256, 256, 0, stream>>>(ctx, woh, bo, out);
}
